// prototypeLayer_50818053047012
// MI455X (gfx1250) — matrix operand order repaired and checked by reading, not compiled, not run on hardware
//
#include <hip/hip_runtime.h>

// Problem constants (match reference): B=16, S=1024, D=768, P=256
#define B_ 16
#define S_ 1024
#define D_ 768
#define P_ 256
#define M_ (B_ * S_)   // 16384 rows of X

typedef __attribute__((ext_vector_type(16))) __bf16 v16bf;  // A/B frag of V_WMMA_F32_16X16X32_BF16
typedef __attribute__((ext_vector_type(8)))  float  v8f;    // 16x16 f32 C/D accumulator

// GEMM tiling: block = 64 M x 256 N (all of P), 1024 threads = 32 waves,
// wave tile = 16 M x 32 N (2 accumulators). K staged in chunks of 32.
#define KC 32
#define BM 64
#define BN 256
#define LROW 40        // padded LDS row stride in bf16 (80B -> distinct bank groups)

// bf16 hi/lo split: x ~= hi + lo, hi/lo products recover ~16-17 mantissa bits.
__device__ __forceinline__ void bf16_split(float x, __bf16& hi, __bf16& lo) {
    hi = (__bf16)x;
    lo = (__bf16)(x - (float)hi);
}

// -------- pass 1: per-row sum of squares (one wave32 per row) --------------
__global__ __launch_bounds__(256) void row_sumsq(const float* __restrict__ src,
                                                 float* __restrict__ dst)
{
    const int gtid = blockIdx.x * 256 + threadIdx.x;
    const int row  = gtid >> 5;          // one wave per row, grid sized exactly
    const int lane = gtid & 31;
    const float4* r4 = (const float4*)(src + (size_t)row * D_);
    float s = 0.f;
#pragma unroll
    for (int i = 0; i < (D_ / 4) / 32; ++i) {   // 6 coalesced float4 loads/lane
        float4 v = r4[lane + i * 32];
        s += v.x * v.x + v.y * v.y + v.z * v.z + v.w * v.w;
    }
#pragma unroll
    for (int off = 16; off > 0; off >>= 1)      // wave32 butterfly reduction
        s += __shfl_xor(s, off, 32);
    if (lane == 0) dst[row] = s;
}

// -------- pass 2: pre-split prototypes into bf16 hi/lo planes (0.77 MB) ----
__global__ __launch_bounds__(256) void split_protos(const float* __restrict__ src,
                                                    __bf16* __restrict__ hi,
                                                    __bf16* __restrict__ lo)
{
    const int i = blockIdx.x * 256 + threadIdx.x;   // exact grid over P_*D_
    __bf16 h, l;
    bf16_split(src[i], h, l);
    hi[i] = h;
    lo[i] = l;
}

// -------- pass 3: bf16x3 WMMA GEMM + distance epilogue ---------------------
// dist[m,n] = x2[m] + p2[n] - 2 * sum_k X[m,k] * Pt[n,k]
__global__ __launch_bounds__(1024) void proto_gemm_dist(
    const float*  __restrict__ X,    // [M_, D_] f32
    const __bf16* __restrict__ Phi,  // [P_, D_] bf16 hi plane
    const __bf16* __restrict__ Plo,  // [P_, D_] bf16 lo plane
    const float*  __restrict__ x2,   // [M_]
    const float*  __restrict__ p2,   // [P_]
    float* __restrict__ out)         // [M_, P_]
{
    // 51.2 KB LDS total: A hi/lo [64][40], B hi/lo [256][40] (bf16)
    __shared__ __bf16 AshHi[BM * LROW];
    __shared__ __bf16 AshLo[BM * LROW];
    __shared__ __bf16 BshHi[BN * LROW];
    __shared__ __bf16 BshLo[BN * LROW];

    const int tid  = threadIdx.x;
    const int lane = tid & 31;
    const int wave = tid >> 5;             // 0..31
    const int wm   = (wave >> 3) * 16;     // 0/16/32/48
    const int wn   = (wave & 7) * 32;      // 0..224
    const int l16  = lane & 15;
    const int lh   = lane >> 4;            // lane half: 0 or 1

    const int mBlk = blockIdx.x * BM;      // N is covered fully (BN == P_)

    // global->reg staging map (1024 threads cover a full K-chunk)
    const int ar = tid >> 4;               // A row   0..63
    const int ac = (tid & 15) * 2;         // A k     0..30 (float2)
    const int br = tid >> 2;               // B proto 0..255
    const int bc = (tid & 3) * 8;          // B k     0..24 (uint4 = 8 bf16)

    float2 aR;
    uint4  bhR, blR;
    {
        aR  = *(const float2*)(X + (mBlk + ar) * D_ + ac);
        bhR = *(const uint4*)(Phi + br * D_ + bc);
        blR = *(const uint4*)(Plo + br * D_ + bc);
    }

    v8f acc[2] = {};                       // in-place accumulators

    union Frag { v16bf v; uint4 q[2]; };

    // stage current regs into LDS (A split to hi/lo on the way in)
    auto stage = [&]() {
        __syncthreads();
        __bf16 h0, l0, h1, l1;
        bf16_split(aR.x, h0, l0);
        bf16_split(aR.y, h1, l1);
        const int ai = ar * LROW + ac;
        AshHi[ai] = h0; AshHi[ai + 1] = h1;
        AshLo[ai] = l0; AshLo[ai + 1] = l1;
        *(uint4*)&BshHi[br * LROW + bc] = bhR;
        *(uint4*)&BshLo[br * LROW + bc] = blR;
        __syncthreads();
    };

    // consume the staged chunk: 6 in-place WMMAs per wave
    auto compute = [&]() {
        // ISA 16-bit A frag: lane -> row l16; low lanes K 0-7 & 16-23,
        // high lanes K 8-15 & 24-31 (two contiguous 16B runs).
        const int kq = lh * 8;
        const __bf16* aHp = &AshHi[(wm + l16) * LROW + kq];
        const __bf16* aLp = &AshLo[(wm + l16) * LROW + kq];
        Frag ah, al;
        ah.q[0] = *(const uint4*)(aHp);      ah.q[1] = *(const uint4*)(aHp + 16);
        al.q[0] = *(const uint4*)(aLp);      al.q[1] = *(const uint4*)(aLp + 16);

        // ISA 16-bit B frag: lane -> col l16; low lanes K 0-7 & 16-23,
        // high lanes K 8-15 & 24-31 (two contiguous 16B runs), same as A.
#pragma unroll
        for (int ni = 0; ni < 2; ++ni) {
            const int n = wn + ni * 16 + l16;
            const __bf16* bHp = &BshHi[n * LROW + lh * 8];
            const __bf16* bLp = &BshLo[n * LROW + lh * 8];
            Frag bh, bl;
            bh.q[0] = *(const uint4*)(bHp);  bh.q[1] = *(const uint4*)(bHp + 16);
            bl.q[0] = *(const uint4*)(bLp);  bl.q[1] = *(const uint4*)(bLp + 16);

            acc[ni] = __builtin_amdgcn_wmma_f32_16x16x32_bf16(
                          false, ah.v, false, bh.v, (short)0, acc[ni], false, false);
            acc[ni] = __builtin_amdgcn_wmma_f32_16x16x32_bf16(
                          false, ah.v, false, bl.v, (short)0, acc[ni], false, false);
            acc[ni] = __builtin_amdgcn_wmma_f32_16x16x32_bf16(
                          false, al.v, false, bh.v, (short)0, acc[ni], false, false);
        }
    };

    // main loop: last iteration peeled so the body is one straight-line block
    // (unconditional prefetch -> stable register assignment, no acc copies).
    const int nIter = D_ / KC;             // 24 K-chunks
    for (int kc = 0; kc < nIter - 1; ++kc) {
        stage();
        const int k0 = (kc + 1) * KC;      // prefetch next chunk (always valid)
        aR  = *(const float2*)(X + (mBlk + ar) * D_ + k0 + ac);
        bhR = *(const uint4*)(Phi + br * D_ + k0 + bc);
        blR = *(const uint4*)(Plo + br * D_ + k0 + bc);
        compute();
    }
    stage();                               // final chunk
    compute();

    // epilogue: D VGPR v holds row (v + 8*lh), col l16 of each 16x16 tile
#pragma unroll
    for (int ni = 0; ni < 2; ++ni) {
        const int n    = wn + ni * 16 + l16;
        const float pv = p2[n];
#pragma unroll
        for (int v = 0; v < 8; ++v) {
            const int m = mBlk + wm + v + 8 * lh;
            out[m * P_ + n] = x2[m] + pv - 2.0f * acc[ni][v];
        }
    }
}

// -------- pass 4: copy prototypes into the tuple tail of d_out -------------
__global__ __launch_bounds__(256) void copy_protos(const float4* __restrict__ src,
                                                   float4* __restrict__ dst)
{
    const int i = blockIdx.x * 256 + threadIdx.x;   // grid sized exactly
    dst[i] = src[i];
}

extern "C" void kernel_launch(void* const* d_in, const int* in_sizes, int n_in,
                              void* d_out, int out_size, void* d_ws, size_t ws_size,
                              hipStream_t stream)
{
    const float* X  = (const float*)d_in[0];   // [16,1024,768] f32
    const float* Pt = (const float*)d_in[1];   // [256,768] f32
    float* out = (float*)d_out;                // distances [M_,P_] then protos [P_,D_]

    // workspace layout (~0.84 MB): x2 | p2 | proto bf16 hi plane | lo plane
    float*  x2  = (float*)d_ws;                          // M_ f32
    float*  p2  = x2 + M_;                               // P_ f32
    __bf16* Phi = (__bf16*)(p2 + P_);                    // P_*D_ bf16 (16B aligned)
    __bf16* Plo = Phi + (size_t)P_ * D_;                 // P_*D_ bf16

    // norms: one wave per row
    row_sumsq<<<dim3((M_ * 32) / 256), 256, 0, stream>>>(X, x2);   // 2048 blocks
    row_sumsq<<<dim3((P_ * 32) / 256), 256, 0, stream>>>(Pt, p2);  //   32 blocks

    // prototype hi/lo planes (split once, reused by every GEMM block)
    split_protos<<<dim3((P_ * D_) / 256), 256, 0, stream>>>(Pt, Phi, Plo);

    // bf16x3 WMMA distance GEMM: 256 blocks x 1024 threads
    proto_gemm_dist<<<dim3(M_ / BM), 1024, 0, stream>>>(X, Phi, Plo, x2, p2, out);

    // tuple tail: prototypes passthrough
    copy_protos<<<dim3((P_ * D_ / 4) / 256), 256, 0, stream>>>(
        (const float4*)Pt, (float4*)out + (size_t)M_ * P_ / 4);
}
